// PVMLayer_42623255446088
// MI455X (gfx1250) — matrix operand order repaired and checked by reading, not compiled, not run on hardware
//
#include <hip/hip_runtime.h>
#include <hip/hip_bf16.h>
#include <stdint.h>

// ---------------------------------------------------------------------------
// PVM / SS2D layer for gfx1250 (MI455X): WMMA f16->f32 GEMMs + selective scan
// with CDNA5 async global->LDS prefetch pipeline (ASYNCcnt).
// dims: B=2, C=256, H=W=48, L=2304, Dm=64, Di=128, R=4, N=16, K=4 dirs, 4 branches
// ---------------------------------------------------------------------------

typedef __attribute__((ext_vector_type(16))) _Float16 v16h;
typedef __attribute__((ext_vector_type(8)))  float    v8f;

#define HH   48
#define WW2  48
#define LL   (HH * WW2)
#define DI   128
#define NST  16
#define CCH  256

union FragH { uint4 q[2]; v16h v; };

// A fragment: 16x32 f16 tile (row-major source, lda in halfs)
// layout: lane<16 -> M=lane, K 0..15; lane>=16 -> M=lane-16, K 16..31 (matches B fragment)
__device__ inline v16h load_frag_a(const _Float16* __restrict__ A, int lda,
                                   int m0, int k0, int lane) {
    FragH f;
    const int hs = lane >> 4;
    const int m  = m0 + (lane & 15);
    const _Float16* base = A + (size_t)m * lda + k0;
    f.q[0] = *(const uint4*)(base + (hs ? 16 : 0));
    f.q[1] = *(const uint4*)(base + (hs ? 24 : 8));
    return f.v;
}

// B fragment: 32x16 f16 tile, weights stored transposed (N-major: Bt[n*ldb + k])
// lane<16 -> N=lane, K 0..15 ; lane>=16 -> N=lane-16, K 16..31
__device__ inline v16h load_frag_b(const _Float16* __restrict__ Bt, int ldb,
                                   int n0, int k0, int lane) {
    FragH f;
    const int hs = lane >> 4;
    const int n  = n0 + (lane & 15);
    const _Float16* base = Bt + (size_t)n * ldb + k0 + (hs ? 16 : 0);
    f.q[0] = *(const uint4*)(base);
    f.q[1] = *(const uint4*)(base + 8);
    return f.v;
}

// Generic tiled WMMA GEMM: C(MxN) = A(MxK) * Bt^T, f16 in / f32 out.
// EPI=0: C[row*ldc + n] (+bias). EPI=1: NCHW scatter out[((row/L)*ldc + n)*L + row%L].
template<int EPI>
__global__ void wmma_gemm_kernel(const _Float16* __restrict__ A, int lda,
                                 const _Float16* __restrict__ Bt,
                                 float* __restrict__ C, int ldc,
                                 const float* __restrict__ bias,
                                 int M, int N, int K, int L)
{
    const int wave = threadIdx.x >> 5;
    const int lane = threadIdx.x & 31;
    const int ntN  = N >> 4;
    const int tile = blockIdx.x * (blockDim.x >> 5) + wave;
    const int total = (M >> 4) * ntN;
    if (tile >= total) return;                  // wave-uniform: EXEC all-ones inside
    const int m0 = (tile / ntN) << 4;
    const int n0 = (tile % ntN) << 4;

    v8f acc = {};
    for (int k0 = 0; k0 < K; k0 += 32) {
        v16h a = load_frag_a(A, lda, m0, k0, lane);
        v16h b = load_frag_b(Bt, K, n0, k0, lane);
        acc = __builtin_amdgcn_wmma_f32_16x16x32_f16(
            /*neg_a=*/false, a, /*neg_b=*/false, b,
            /*c_mod=*/(short)0, acc, /*reuse_a=*/false, /*reuse_b=*/false);
    }

    const int n     = n0 + (lane & 15);
    const int mbase = m0 + (lane >> 4) * 8;
    const float bv  = bias ? bias[n] : 0.f;
#pragma unroll
    for (int r = 0; r < 8; ++r) {
        const int row = mbase + r;
        const float v = acc[r] + bv;
        if (EPI == 0) {
            C[(size_t)row * ldc + n] = v;
        } else {
            const int b_ = row / L, l_ = row % L;
            C[((size_t)b_ * ldc + n) * L + l_] = v;
        }
    }
}

// Convert weights to f16 (transposed to N-major where needed)
__global__ void prep_weights_kernel(const float* __restrict__ in_W,   // (64,128)
                                    const float* __restrict__ xp_W,   // (4,36,128)
                                    const float* __restrict__ out_W,  // (128,64)
                                    const float* __restrict__ proj_W, // (256,256)
                                    _Float16* __restrict__ in_Wt,     // (128,64)
                                    _Float16* __restrict__ xp_h,      // (144,128)
                                    _Float16* __restrict__ out_Wt,    // (64,128)
                                    _Float16* __restrict__ proj_Wt)   // (256,256)
{
    const int i = blockIdx.x * blockDim.x + threadIdx.x;
    if (i < 128 * 64)  { int n = i / 64,  k = i % 64;  in_Wt[i]  = (_Float16)in_W[k * 128 + n]; }
    if (i < 144 * 128) {                               xp_h[i]   = (_Float16)xp_W[i]; }
    if (i < 64 * 128)  { int n = i / 128, k = i % 128; out_Wt[i] = (_Float16)out_W[k * 64 + n]; }
    if (i < 256 * 256) { int n = i / 256, k = i % 256; proj_Wt[i]= (_Float16)proj_W[k * 256 + n]; }
}

// LayerNorm over C=256 of NCHW input -> channel-last f32 + f16. One wave / pixel.
__global__ void ln_x_kernel(const float* __restrict__ x,
                            const float* __restrict__ g, const float* __restrict__ b,
                            float* __restrict__ xn, _Float16* __restrict__ xn_h, int nPix)
{
    const int pix = blockIdx.x * (blockDim.x >> 5) + (threadIdx.x >> 5);
    if (pix >= nPix) return;
    const int lane = threadIdx.x & 31;
    const int bb = pix / LL, p = pix % LL;
    const float* xb = x + (size_t)bb * CCH * LL + p;
    float v[8], s = 0.f;
#pragma unroll
    for (int j = 0; j < 8; ++j) { v[j] = xb[(size_t)(lane * 8 + j) * LL]; s += v[j]; }
#pragma unroll
    for (int o = 16; o; o >>= 1) s += __shfl_xor(s, o);
    const float mean = s * (1.f / 256.f);
    float vs = 0.f;
#pragma unroll
    for (int j = 0; j < 8; ++j) { float d = v[j] - mean; vs += d * d; }
#pragma unroll
    for (int o = 16; o; o >>= 1) vs += __shfl_xor(vs, o);
    const float inv = rsqrtf(vs * (1.f / 256.f) + 1e-5f);
#pragma unroll
    for (int j = 0; j < 8; ++j) {
        const int c = lane * 8 + j;
        const float o = (v[j] - mean) * inv * g[c] + b[c];
        xn[(size_t)pix * CCH + c]   = o;
        xn_h[(size_t)pix * CCH + c] = (_Float16)o;
    }
}

// depthwise 3x3 SAME + bias + SiLU ; also zero y_pix accumulator
__global__ void conv_silu_kernel(const float* __restrict__ xin,
                                 const float* __restrict__ cw, const float* __restrict__ cb,
                                 float* __restrict__ u, _Float16* __restrict__ u_h,
                                 float* __restrict__ y_pix)
{
    const int c  = threadIdx.x;          // 128 channels
    const int gp = blockIdx.x;           // (branch*B + b)*L + p
    const int bb = gp / LL, p = gp % LL;
    const int h = p / WW2, w = p % WW2;
    const float* base = xin + (size_t)bb * LL * DI + c;
    float acc = cb[c];
#pragma unroll
    for (int kh = 0; kh < 3; ++kh) {
        const int hh = h + kh - 1;
        if ((unsigned)hh >= HH) continue;
#pragma unroll
        for (int kw = 0; kw < 3; ++kw) {
            const int ww = w + kw - 1;
            if ((unsigned)ww >= WW2) continue;
            acc += base[(size_t)(hh * WW2 + ww) * DI] * cw[c * 9 + kh * 3 + kw];
        }
    }
    const float sv = acc / (1.f + __expf(-acc));   // SiLU
    const size_t idx = (size_t)gp * DI + c;
    u[idx] = sv; u_h[idx] = (_Float16)sv; y_pix[idx] = 0.f;
}

// ---------------------------------------------------------------------------
// Selective scan: one block per (branch,batch,dir). 512 thr: t -> d=t>>2, 4 states.
// Depth-2 async global->LDS prefetch pipeline (4 buffers, one barrier/step):
//   buffer layout per step: [0..35] = dbl segment (dt|B|C), [36..163] = u row.
//   wave0 lanes 0..8 fetch dbl seg (9 x b128); wave1 lanes 0..31 fetch u row.
//   step l issues load(l+2), waits ASYNCcnt<=2 (=> load(l) complete), barrier.
// ---------------------------------------------------------------------------
__global__ void scan_kernel(const float* __restrict__ dbl,    // (8B*L, 144)
                            const float* __restrict__ u,      // (8B*L, 128)
                            const float* __restrict__ dtW,    // (4,128,4)
                            const float* __restrict__ dtb,    // (4,128)
                            const float* __restrict__ A_logs, // (4*128,16)
                            const float* __restrict__ Ds,     // (4*128,)
                            float* __restrict__ y_pix)        // (8B*L, 128)
{
    __shared__ float pref[4][164];       // 4 buffers x (36 dbl + 128 u), 16B aligned rows
    const int s  = blockIdx.x;
    const int bb = s >> 2;               // branch*B + b
    const int k  = s & 3;                // scan direction
    const size_t rowbase = (size_t)bb * LL;
    const int t    = threadIdx.x;
    const int wv   = t >> 5;
    const int lane = t & 31;
    const int d    = t >> 2;
    const int nq   = t & 3;

    float Aj[4], hj[4] = {0.f, 0.f, 0.f, 0.f};
#pragma unroll
    for (int j = 0; j < 4; ++j)
        Aj[j] = -__expf(A_logs[(size_t)(k * DI + d) * NST + (4 * nq + j)]);
    float dw[4];
#pragma unroll
    for (int r = 0; r < 4; ++r) dw[r] = dtW[(size_t)(k * DI + d) * 4 + r];
    const float db   = dtb[k * DI + d];
    const float dval = Ds[k * DI + d];

    auto pix_of = [&](int l) -> int {
        if (k == 0) return l;
        if (k == 2) return LL - 1 - l;
        const int ll = (k == 1) ? l : (LL - 1 - l);
        return (ll % HH) * WW2 + (ll / HH);      // col-major -> row-major pixel
    };

    auto issue = [&](int l, int buf) {
        const int lc = (l < LL) ? l : (LL - 1);  // clamp tail: keep count uniform
        const int p  = pix_of(lc);
        if (wv == 0) {
            if (lane < 9) {
                const float* gp = dbl + (rowbase + p) * 144 + k * 36 + lane * 4;
                const unsigned lo = (unsigned)(uintptr_t)(&pref[buf][lane * 4]);
                asm volatile("global_load_async_to_lds_b128 %0, %1, off"
                             :: "v"(lo), "v"((unsigned long long)(uintptr_t)gp)
                             : "memory");
            }
        } else if (wv == 1) {
            const float* gp = u + (rowbase + p) * DI + lane * 4;
            const unsigned lo = (unsigned)(uintptr_t)(&pref[buf][36 + lane * 4]);
            asm volatile("global_load_async_to_lds_b128 %0, %1, off"
                         :: "v"(lo), "v"((unsigned long long)(uintptr_t)gp)
                         : "memory");
        }
    };

    issue(0, 0);
    issue(1, 1);

    for (int l = 0; l < LL; ++l) {
        const int buf = l & 3;
        issue(l + 2, (l + 2) & 3);
        if (wv < 2)
            asm volatile("s_wait_asynccnt 2" ::: "memory");   // load(l) complete
        __syncthreads();                                      // publish buf to all waves

        const float* sg = pref[buf];
        float x = db;
#pragma unroll
        for (int r = 0; r < 4; ++r) x += dw[r] * sg[r];
        const float delta = fmaxf(x, 0.f) + log1pf(__expf(-fabsf(x)));  // softplus
        const float u_d = sg[36 + d];
        const float du  = delta * u_d;
        float part = 0.f;
#pragma unroll
        for (int j = 0; j < 4; ++j) {
            const int n = 4 * nq + j;
            hj[j] = hj[j] * __expf(delta * Aj[j]) + du * sg[4 + n];
            part += hj[j] * sg[20 + n];
        }
        part += __shfl_xor(part, 1);
        part += __shfl_xor(part, 2);
        if (nq == 0)
            atomicAdd(&y_pix[(rowbase + pix_of(l)) * DI + d], part + u_d * dval);
    }
    if (wv < 2)
        asm volatile("s_wait_asynccnt 0" ::: "memory");       // drain before exit
}

// LN over Di=128 (onorm) -> f16 for out_proj. One wave / row.
__global__ void onorm_kernel(const float* __restrict__ y_pix,
                             const float* __restrict__ g, const float* __restrict__ b,
                             _Float16* __restrict__ yln_h, int nRows)
{
    const int row = blockIdx.x * (blockDim.x >> 5) + (threadIdx.x >> 5);
    if (row >= nRows) return;
    const int lane = threadIdx.x & 31;
    const float4 v = ((const float4*)(y_pix + (size_t)row * DI))[lane];
    float s = v.x + v.y + v.z + v.w;
#pragma unroll
    for (int o = 16; o; o >>= 1) s += __shfl_xor(s, o);
    const float mean = s * (1.f / 128.f);
    float vs = (v.x-mean)*(v.x-mean) + (v.y-mean)*(v.y-mean)
             + (v.z-mean)*(v.z-mean) + (v.w-mean)*(v.w-mean);
#pragma unroll
    for (int o = 16; o; o >>= 1) vs += __shfl_xor(vs, o);
    const float inv = rsqrtf(vs * (1.f / 128.f) + 1e-5f);
    const float e[4] = {v.x, v.y, v.z, v.w};
#pragma unroll
    for (int j = 0; j < 4; ++j) {
        const int c = lane * 4 + j;
        yln_h[(size_t)row * DI + c] = (_Float16)((e[j] - mean) * inv * g[c] + b[c]);
    }
}

// branch outputs + skip, then LN(ln_g/ln_b) over 256 -> f16 for final proj
__global__ void final_pre_kernel(const float* __restrict__ yo,   // (4B*L, 64)
                                 const float* __restrict__ xn,   // (B*L, 256)
                                 const float* __restrict__ skip,
                                 const float* __restrict__ g, const float* __restrict__ b,
                                 _Float16* __restrict__ z_h, int Bsz, int nPix)
{
    const int pix = blockIdx.x * (blockDim.x >> 5) + (threadIdx.x >> 5);
    if (pix >= nPix) return;
    const int lane = threadIdx.x & 31;
    const int bi = pix / LL, l = pix % LL;
    const float sk = skip[0];
    float v[8], s = 0.f;
#pragma unroll
    for (int j = 0; j < 8; ++j) {
        const int c = lane * 8 + j, br = c >> 6, jj = c & 63;
        v[j] = yo[((size_t)(br * Bsz + bi) * LL + l) * 64 + jj]
             + sk * xn[(size_t)pix * CCH + c];
        s += v[j];
    }
#pragma unroll
    for (int o = 16; o; o >>= 1) s += __shfl_xor(s, o);
    const float mean = s * (1.f / 256.f);
    float vs = 0.f;
#pragma unroll
    for (int j = 0; j < 8; ++j) { float d = v[j] - mean; vs += d * d; }
#pragma unroll
    for (int o = 16; o; o >>= 1) vs += __shfl_xor(vs, o);
    const float inv = rsqrtf(vs * (1.f / 256.f) + 1e-5f);
#pragma unroll
    for (int j = 0; j < 8; ++j) {
        const int c = lane * 8 + j;
        z_h[(size_t)pix * CCH + c] = (_Float16)((v[j] - mean) * inv * g[c] + b[c]);
    }
}

// ---------------------------------------------------------------------------
extern "C" void kernel_launch(void* const* d_in, const int* in_sizes, int n_in,
                              void* d_out, int out_size, void* d_ws, size_t ws_size,
                              hipStream_t stream) {
    const float* x        = (const float*)d_in[0];
    const float* ln_g     = (const float*)d_in[1];
    const float* ln_b     = (const float*)d_in[2];
    const float* skip     = (const float*)d_in[3];
    const float* proj_W   = (const float*)d_in[4];
    const float* proj_b   = (const float*)d_in[5];
    const float* in_W     = (const float*)d_in[6];
    const float* conv_W   = (const float*)d_in[7];
    const float* conv_b   = (const float*)d_in[8];
    const float* x_proj_W = (const float*)d_in[9];
    const float* dt_W     = (const float*)d_in[10];
    const float* dt_b     = (const float*)d_in[11];
    const float* A_logs   = (const float*)d_in[12];
    const float* Ds       = (const float*)d_in[13];
    const float* onorm_g  = (const float*)d_in[14];
    const float* onorm_b  = (const float*)d_in[15];
    const float* out_W    = (const float*)d_in[16];
    float* out = (float*)d_out;

    const int B    = in_sizes[0] / (CCH * LL);   // 2
    const int nPix = B * LL;                     // 4608
    const int R8   = 4 * B * LL;                 // 18432 rows (branch-major)

    char* w = (char*)d_ws;
    auto carve = [&](size_t bytes) -> void* {
        void* r = (void*)w; w += (bytes + 255) & ~(size_t)255; return r;
    };
    float*     xn      = (float*)    carve((size_t)nPix * CCH * 4);
    _Float16*  xn_h    = (_Float16*) carve((size_t)nPix * CCH * 2);
    float*     xin     = (float*)    carve((size_t)R8 * DI * 4);
    float*     u       = (float*)    carve((size_t)R8 * DI * 4);
    _Float16*  u_h     = (_Float16*) carve((size_t)R8 * DI * 2);
    float*     dbl     = (float*)    carve((size_t)R8 * 144 * 4);
    float*     y_pix   = (float*)    carve((size_t)R8 * DI * 4);
    _Float16*  yln_h   = (_Float16*) carve((size_t)R8 * DI * 2);
    float*     yo      = (float*)    carve((size_t)R8 * 64 * 4);
    _Float16*  z_h     = (_Float16*) carve((size_t)nPix * CCH * 2);
    _Float16*  in_Wt   = (_Float16*) carve(128 * 64 * 2);
    _Float16*  xp_h    = (_Float16*) carve(144 * 128 * 2);
    _Float16*  out_Wt  = (_Float16*) carve(64 * 128 * 2);
    _Float16*  proj_Wt = (_Float16*) carve(256 * 256 * 2);

    // 1. weight conversion (f16, N-major)
    prep_weights_kernel<<<256, 256, 0, stream>>>(in_W, x_proj_W, out_W, proj_W,
                                                 in_Wt, xp_h, out_Wt, proj_Wt);
    // 2. LN over C
    ln_x_kernel<<<(nPix + 7) / 8, 256, 0, stream>>>(x, ln_g, ln_b, xn, xn_h, nPix);
    // 3. in_proj per branch: (nPix x 64) @ (64 x 128)
    for (int br = 0; br < 4; ++br) {
        const int tiles = (nPix / 16) * (DI / 16);
        wmma_gemm_kernel<0><<<(tiles + 3) / 4, 128, 0, stream>>>(
            xn_h + br * 64, CCH, in_Wt, xin + (size_t)br * nPix * DI, DI,
            nullptr, nPix, DI, 64, LL);
    }
    // 4. depthwise conv + SiLU (+ zero accumulator)
    conv_silu_kernel<<<R8, DI, 0, stream>>>(xin, conv_W, conv_b, u, u_h, y_pix);
    // 5. x_proj for all 4 dirs at once: (R8 x 128) @ (128 x 144)
    {
        const int tiles = (R8 / 16) * (144 / 16);
        wmma_gemm_kernel<0><<<(tiles + 3) / 4, 128, 0, stream>>>(
            u_h, DI, xp_h, dbl, 144, nullptr, R8, 144, DI, LL);
    }
    // 6. selective scans: 32 blocks of 512 threads, async-prefetch pipeline
    scan_kernel<<<4 * B * 4, 512, 0, stream>>>(dbl, u, dt_W, dt_b, A_logs, Ds, y_pix);
    // 7. out-norm LN over 128
    onorm_kernel<<<(R8 + 7) / 8, 256, 0, stream>>>(y_pix, onorm_g, onorm_b, yln_h, R8);
    // 8. out_proj: (R8 x 128) @ (128 x 64)
    {
        const int tiles = (R8 / 16) * (64 / 16);
        wmma_gemm_kernel<0><<<(tiles + 3) / 4, 128, 0, stream>>>(
            yln_h, DI, out_Wt, yo, 64, nullptr, R8, 64, DI, LL);
    }
    // 9. concat + skip + LN
    final_pre_kernel<<<(nPix + 7) / 8, 256, 0, stream>>>(yo, xn, skip, ln_g, ln_b,
                                                         z_h, B, nPix);
    // 10. final proj with bias, NCHW scatter epilogue
    {
        const int tiles = (nPix / 16) * (CCH / 16);
        wmma_gemm_kernel<1><<<(tiles + 3) / 4, 128, 0, stream>>>(
            z_h, CCH, proj_Wt, out, CCH, proj_b, nPix, CCH, CCH, LL);
    }
}
